// TripletLossV2_29222957482889
// MI455X (gfx1250) — matrix operand order repaired and checked by reading, not compiled, not run on hardware
//
#include <hip/hip_runtime.h>
#include <hip/hip_bf16.h>

// Problem constants (from reference)
#define B_ 2
#define C_ 128
#define H_ 80
#define W_ 80
#define N_ (H_ * W_)   // 6400 pixels per image

// ---- CDNA5 WMMA / TDM types ----
typedef __attribute__((ext_vector_type(16))) __bf16 v16bf;
typedef __attribute__((ext_vector_type(8)))  float  v8f;
typedef __attribute__((ext_vector_type(4)))  unsigned int u32x4;
typedef __attribute__((ext_vector_type(8)))  int i32x8;
typedef __attribute__((ext_vector_type(4)))  int i32x4;

union BfPack { u32x4 q[2]; v16bf v; };

#if defined(__has_builtin)
#if __has_builtin(__builtin_amdgcn_tensor_load_to_lds) && \
    __has_builtin(__builtin_amdgcn_s_wait_tensorcnt)
#define USE_TDM 1
#endif
#endif
#ifndef USE_TDM
#define USE_TDM 0
#endif

// TDM staging geometry: 64 columns (= 64 bf16 feature vectors of 256B) per DMA,
// written to LDS with a 16B pad after every 256B row (row pitch 272B) so that
// ds_load_b128 fragment reads hit the minimal 2-cycle bank pattern.
#define MT_COLS   64
#define ROW_PITCH 272
#define TILE_LDS  (MT_COLS * ROW_PITCH)   // 17408 B per buffer
#define NBUF      3
#define NMT       (N_ / (16 * 4))         // 100 macro column tiles (64 cols each)

// round-to-nearest-even f32 -> bf16 (inputs are finite)
__device__ __forceinline__ unsigned short f2bf(float f) {
  unsigned u = __float_as_uint(f);
  u += 0x7FFFu + ((u >> 16) & 1u);
  return (unsigned short)(u >> 16);
}

#if USE_TDM
// Low 32 bits of a generic LDS pointer == logical LDS byte address
// (SHARED_BASE aperture lives entirely in the high dword).
__device__ __forceinline__ unsigned lds_offset_of(const void* p) {
  return (unsigned)(unsigned long long)p;
}

// Issue one TDM load: 64 rows x 256B, contiguous in global, padded into LDS.
// D# per CDNA5 ISA (group0 128b, group1 256b).
__device__ __forceinline__ void tdm_load_tile(const unsigned short* gsrc,
                                              unsigned lds_addr) {
  unsigned long long ga = (unsigned long long)gsrc;
  u32x4 g0;
  g0[0] = 1u;                                        // count=1, user descriptor
  g0[1] = lds_addr;                                  // lds_addr (bytes)
  g0[2] = (unsigned)(ga & 0xFFFFFFFFull);            // global_addr[31:0]
  g0[3] = ((unsigned)(ga >> 32) & 0x01FFFFFFu)       // global_addr[56:32]
        | 0x80000000u;                               // type=2 ("image")
  i32x8 g1;
  g1[0] = (int)((1u << 20)                           // pad_enable
              | (5u << 22)                           // pad_interval: 64 DWORDs
              | (3u << 25));                         // pad_amount:   4 DWORDs
                                                     // wg_mask=0, data_size=1B
  g1[1] = (int)(256u << 16);                         // tensor_dim0 = 256 B
  g1[2] = (int)((unsigned)MT_COLS << 16);            // tensor_dim1 = 64 rows
  g1[3] = (int)(256u << 16);                         // tile_dim0 = 256 B
  g1[4] = MT_COLS;                                   // tile_dim1 = 64 rows
  g1[5] = 256;                                       // tensor_dim0_stride
  g1[6] = 0;
  g1[7] = 0;
  i32x4 z4 = {0, 0, 0, 0};
#if __clang_major__ >= 23
  i32x8 z8 = {0, 0, 0, 0, 0, 0, 0, 0};
  __builtin_amdgcn_tensor_load_to_lds(g0, g1, z4, z4, z8, 0);
#else
  __builtin_amdgcn_tensor_load_to_lds(g0, g1, z4, z4, 0);
#endif
}
#endif  // USE_TDM

// -----------------------------------------------------------------------------
// Kernel 1: per-pixel preprocessing (flow -> correspondence, f32 pos_dist,
// bf16 transposed feature copies [n][c], 256B per vector).
// -----------------------------------------------------------------------------
__global__ __launch_bounds__(256)
void prep_kernel(const float* __restrict__ feat1,
                 const float* __restrict__ feat2,
                 const float* __restrict__ aflow,
                 unsigned short* __restrict__ f1T,
                 unsigned short* __restrict__ f2T,
                 unsigned short* __restrict__ f2mT,
                 int*   __restrict__ x2a,
                 int*   __restrict__ y2a,
                 float* __restrict__ validf,
                 float* __restrict__ posd) {
  int t = blockIdx.x * blockDim.x + threadIdx.x;
  if (t >= B_ * N_) return;
  int b = t / N_;
  int n = t - b * N_;

  float afx = aflow[((size_t)b * 2 + 0) * N_ + n];
  float afy = aflow[((size_t)b * 2 + 1) * N_ + n];
  int x2 = (int)floorf(afx + 0.5f);
  int y2 = (int)floorf(afy + 0.5f);
  int valid = (x2 >= 0) & (y2 >= 0) & (x2 < W_) & (y2 < H_);
  int x2c = x2 < 0 ? 0 : (x2 > W_ - 1 ? W_ - 1 : x2);
  int y2c = y2 < 0 ? 0 : (y2 > H_ - 1 ? H_ - 1 : y2);
  int m2  = y2c * W_ + x2c;

  const float* f1 = feat1 + (size_t)b * C_ * N_;
  const float* f2 = feat2 + (size_t)b * C_ * N_;
  unsigned short* o1 = f1T  + (size_t)b * N_ * C_ + (size_t)n * C_;
  unsigned short* o2 = f2T  + (size_t)b * N_ * C_ + (size_t)n * C_;
  unsigned short* o3 = f2mT + (size_t)b * N_ * C_ + (size_t)n * C_;

  float dot = 0.f;
  #pragma unroll 4
  for (int c = 0; c < C_; ++c) {
    float v1 = f1[(size_t)c * N_ + n];
    float v2 = f2[(size_t)c * N_ + n];
    float vg = f2[(size_t)c * N_ + m2];
    dot = fmaf(v1, vg, dot);
    o1[c] = f2bf(v1);
    o2[c] = f2bf(v2);
    o3[c] = f2bf(vg);
  }
  x2a[t]    = x2c;
  y2a[t]    = y2c;
  validf[t] = valid ? 1.f : 0.f;
  posd[t]   = 2.f - 2.f * dot;
}

// -----------------------------------------------------------------------------
// Kernel 2: fused WMMA GEMM + penalty + row-min.
//  8 waves/block share (batch, kind) and the full column sweep; B tiles are
//  staged to LDS by the Tensor Data Mover (wave 0 issues, TENSORcnt tracked,
//  triple-buffered, 2 macro-tiles of prefetch depth).
// -----------------------------------------------------------------------------
__global__ __launch_bounds__(256)
void gemmmin_kernel(const unsigned short* __restrict__ f1T,
                    const unsigned short* __restrict__ f2T,
                    const unsigned short* __restrict__ f2mT,
                    const int* __restrict__ x2a,
                    const int* __restrict__ y2a,
                    float* __restrict__ negmin) {
  const int wave = threadIdx.x >> 5;
  const int lane = threadIdx.x & 31;
  const int task = blockIdx.x * 8 + wave;      // 0..1599
  const int b    = task / 800;
  const int kind = (task / 400) & 1;           // 0: neg1, 1: neg2
  const int rt   = task % 400;
  const int rowBase = rt * 16;
  const int half = lane >> 4;
  const int lm   = lane & 15;

#if USE_TDM
  __shared__ __align__(16) unsigned char ldsB[NBUF * TILE_LDS];
#endif

  const unsigned short* Amat = (kind == 0 ? f1T : f2mT) + (size_t)b * N_ * C_;
  const unsigned short* Bmat = (kind == 0 ? f2T : f1T)  + (size_t)b * N_ * C_;

  // ---- A fragments: lane=row lm, half selects the 16-wide contiguous
  //      K slab; 4 x v16bf = 32 VGPRs, resident all sweep.
  v16bf a[4];
  {
    const unsigned short* Ap = Amat + (size_t)(rowBase + lm) * C_;
    #pragma unroll
    for (int cc = 0; cc < 4; ++cc) {
      int k0 = 32 * cc + 16 * half;
      BfPack p;
      p.q[0] = *(const u32x4*)(Ap + k0);
      p.q[1] = *(const u32x4*)(Ap + k0 + 8);
      a[cc] = p.v;
    }
  }

  // Row geometry for the 8 accumulator rows this lane owns (M = v + 8*half).
  int rx[8], ry[8];
  {
    int rb = rowBase + 8 * half;
    if (kind == 0) {
      #pragma unroll
      for (int v = 0; v < 8; ++v) {
        rx[v] = x2a[b * N_ + rb + v];
        ry[v] = y2a[b * N_ + rb + v];
      }
    } else {
      int cy = rb / W_, cx = rb % W_;   // 16 | rowBase, 16 | 80 => no wrap
      #pragma unroll
      for (int v = 0; v < 8; ++v) { rx[v] = cx + v; ry[v] = cy; }
    }
  }
  // neg1: sqrt(d2+1e-4)<=5 => d2<=24 (d2==25 escapes); neg2: d2<=25.
  const int thr = (kind == 0) ? 24 : 25;

  float rowmin[8];
  #pragma unroll
  for (int v = 0; v < 8; ++v) rowmin[v] = 1e30f;

  auto epilogue = [&](const v8f& acc, int colBase) {
    const int colY = colBase / W_;
    const int colX = colBase % W_ + lm;
    #pragma unroll
    for (int v = 0; v < 8; ++v) {
      int dx = rx[v] - colX;
      int dy = ry[v] - colY;
      int d2 = dx * dx + dy * dy;
      float base = (d2 <= thr) ? 12.f : 2.f;          // (2-2*dot)+10 fused
      float val  = fmaf(-2.f, acc[v], base);
      rowmin[v]  = fminf(rowmin[v], val);
    }
  };

#if USE_TDM
  if (wave == 0) {
    tdm_load_tile(Bmat,                         lds_offset_of(&ldsB[0]));
    tdm_load_tile(Bmat + (size_t)MT_COLS * C_,  lds_offset_of(&ldsB[TILE_LDS]));
  }
  for (int mt = 0; mt < NMT; ++mt) {
    // Barrier #1: every wave finished reading buf[(mt-1)%3] (the buffer the
    // next DMA will overwrite) before wave 0 issues it.
    __syncthreads();
    if (wave == 0) {
      if (mt + 2 < NMT)
        tdm_load_tile(Bmat + (size_t)(mt + 2) * MT_COLS * C_,
                      lds_offset_of(&ldsB[((mt + 2) % NBUF) * TILE_LDS]));
      // TDM completes in order: <=2 outstanding ==> tile mt is in LDS.
      if (mt < NMT - 2)       __builtin_amdgcn_s_wait_tensorcnt(2);
      else if (mt == NMT - 2) __builtin_amdgcn_s_wait_tensorcnt(1);
      else                    __builtin_amdgcn_s_wait_tensorcnt(0);
    }
    // Barrier #2: publish buf[mt%3] to all 8 waves.
    __syncthreads();

    const unsigned char* lbase = &ldsB[(mt % NBUF) * TILE_LDS];
    #pragma unroll
    for (int st = 0; st < 4; ++st) {
      // Fragment (contiguous K order, same as A): lane=column lm, half selects the 16-wide
      // K slab; row pitch 272B from the TDM pad => 2-way-minimal banks.
      const unsigned char* lrow = lbase + (st * 16 + lm) * ROW_PITCH + 32 * half;
      v8f acc = {0.f, 0.f, 0.f, 0.f, 0.f, 0.f, 0.f, 0.f};
      #pragma unroll
      for (int cc = 0; cc < 4; ++cc) {
        BfPack p;
        p.q[0] = *(const u32x4*)(lrow + 64 * cc);
        p.q[1] = *(const u32x4*)(lrow + 64 * cc + 16);
        acc = __builtin_amdgcn_wmma_f32_16x16x32_bf16(
            false, a[cc], false, p.v, (short)0, acc, false, false);
      }
      epilogue(acc, mt * MT_COLS + st * 16);
    }
  }
#else
  // Fallback: per-lane direct global B fragments (proven baseline path).
  for (int ct = 0; ct < 400; ++ct) {
    const int colBase = ct * 16;
    const unsigned short* Bp = Bmat + (size_t)(colBase + lm) * C_;
    if (ct + 1 < 400) __builtin_prefetch(Bp + 16 * C_, 0, 3);
    v8f acc = {0.f, 0.f, 0.f, 0.f, 0.f, 0.f, 0.f, 0.f};
    #pragma unroll
    for (int cc = 0; cc < 4; ++cc) {
      int k0 = 32 * cc + 16 * half;
      BfPack p;
      p.q[0] = *(const u32x4*)(Bp + k0);
      p.q[1] = *(const u32x4*)(Bp + k0 + 8);
      acc = __builtin_amdgcn_wmma_f32_16x16x32_bf16(
          false, a[cc], false, p.v, (short)0, acc, false, false);
    }
    epilogue(acc, colBase);
  }
#endif

  // Min across the 16 lanes sharing each row (butterfly stays in-group).
  #pragma unroll
  for (int v = 0; v < 8; ++v) {
    float m = rowmin[v];
    m = fminf(m, __shfl_xor(m, 1, 32));
    m = fminf(m, __shfl_xor(m, 2, 32));
    m = fminf(m, __shfl_xor(m, 4, 32));
    m = fminf(m, __shfl_xor(m, 8, 32));
    rowmin[v] = m;
  }
  if (lm == 0) {  // lanes 0 (rows +0..7) and 16 (rows +8..15)
    float* out = negmin + (size_t)kind * B_ * N_ + (size_t)b * N_
               + rowBase + 8 * half;
    #pragma unroll
    for (int v = 0; v < 8; ++v) out[v] = rowmin[v];
  }
}

// -----------------------------------------------------------------------------
// Kernel 3: deterministic reduction to the scalar loss.
// -----------------------------------------------------------------------------
__global__ __launch_bounds__(256)
void finalize_kernel(const float* __restrict__ negmin,
                     const float* __restrict__ posd,
                     const float* __restrict__ validf,
                     float* __restrict__ out) {
  __shared__ float ssum[256];
  __shared__ float scnt[256];
  int tid = threadIdx.x;
  float total = 0.f;
  for (int b = 0; b < B_; ++b) {
    float s = 0.f, c = 0.f;
    for (int i = tid; i < N_; i += 256) {
      float neg  = fminf(negmin[0 * B_ * N_ + b * N_ + i],
                         negmin[1 * B_ * N_ + b * N_ + i]);
      float diff = fmaxf(0.f, 1.f + posd[b * N_ + i] - neg);
      float w    = validf[b * N_ + i];
      s += diff * w;
      c += w;
    }
    ssum[tid] = s; scnt[tid] = c;
    __syncthreads();
    for (int o = 128; o > 0; o >>= 1) {
      if (tid < o) { ssum[tid] += ssum[tid + o]; scnt[tid] += scnt[tid + o]; }
      __syncthreads();
    }
    if (tid == 0) total += ssum[0] / fmaxf(scnt[0], 1.f);
    __syncthreads();
  }
  if (tid == 0) out[0] = total / (float)B_;
}

// -----------------------------------------------------------------------------
extern "C" void kernel_launch(void* const* d_in, const int* in_sizes, int n_in,
                              void* d_out, int out_size, void* d_ws, size_t ws_size,
                              hipStream_t stream) {
  const float* feat1 = (const float*)d_in[0];
  const float* feat2 = (const float*)d_in[1];
  const float* aflow = (const float*)d_in[2];
  float* out = (float*)d_out;

  char* ws = (char*)d_ws;
  const size_t featBytes = (size_t)B_ * N_ * C_ * sizeof(unsigned short);
  unsigned short* f1T  = (unsigned short*)ws; ws += featBytes;
  unsigned short* f2T  = (unsigned short*)ws; ws += featBytes;
  unsigned short* f2mT = (unsigned short*)ws; ws += featBytes;
  int*   x2a    = (int*)ws;   ws += (size_t)B_ * N_ * sizeof(int);
  int*   y2a    = (int*)ws;   ws += (size_t)B_ * N_ * sizeof(int);
  float* validf = (float*)ws; ws += (size_t)B_ * N_ * sizeof(float);
  float* posd   = (float*)ws; ws += (size_t)B_ * N_ * sizeof(float);
  float* negmin = (float*)ws; ws += (size_t)2 * B_ * N_ * sizeof(float);

  hipLaunchKernelGGL(prep_kernel, dim3((B_ * N_ + 255) / 256), dim3(256), 0, stream,
                     feat1, feat2, aflow, f1T, f2T, f2mT, x2a, y2a, validf, posd);
  hipLaunchKernelGGL(gemmmin_kernel, dim3(200), dim3(256), 0, stream,
                     f1T, f2T, f2mT, x2a, y2a, negmin);
  hipLaunchKernelGGL(finalize_kernel, dim3(1), dim3(256), 0, stream,
                     negmin, posd, validf, out);
}
